// Mamba_44470091383264
// MI455X (gfx1250) — matrix operand order repaired and checked by reading, not compiled, not run on hardware
//
#include <hip/hip_runtime.h>
#include <cstdint>
#include <cstddef>

// Problem constants (from reference): B, L, DM, ED, N, R, K
#define B_   2
#define L_   1024
#define DM_  1024
#define ED_  2048
#define NS_  16      // state size N
#define RR_  64      // dt rank
#define FF_  96      // R + 2N
#define FP_  128     // padded rows for x_proj_w (OOB-safe async tiles)
#define NC_  8       // scan chunks along L
#define LC_  128     // chunk length (L_/NC_)

typedef __attribute__((ext_vector_type(16))) __bf16 v16bf;
typedef __attribute__((ext_vector_type(8)))  __bf16 v8bf;
typedef __attribute__((ext_vector_type(8)))  float  v8f;
typedef __attribute__((ext_vector_type(4)))  int    v4i;

__device__ __forceinline__ float sigmoidf_(float x) { return 1.0f / (1.0f + __expf(-x)); }
__device__ __forceinline__ float softplusf_(float x) {
  return (x > 20.0f) ? x : __logf(1.0f + __expf(x));
}

// ---------------------------------------------------------------------------
// Async global->LDS tile copy (gfx1250 GLOBAL_LOAD_ASYNC_TO_LDS_B128, ASYNCcnt).
// Builtin signature (probe-verified): (int4 AS1*, int4 AS3*, Imm, Imm).
// Falls back to a b128 register round-trip if the builtin is not declared.
// ---------------------------------------------------------------------------
#if defined(__has_builtin)
#  if __has_builtin(__builtin_amdgcn_global_load_async_to_lds_b128)
#    define ASYNC_TILE 1
#  endif
#endif
#ifndef ASYNC_TILE
#  define ASYNC_TILE 0
#endif

__device__ __forceinline__ void copy16(const __bf16* g, __bf16* l)
{
#if ASYNC_TILE
  // Generic LDS pointer low 32 bits == LDS offset (ISA 10.2 aperture rules).
  __builtin_amdgcn_global_load_async_to_lds_b128(
      (__attribute__((address_space(1))) v4i*)(uintptr_t)g,
      (__attribute__((address_space(3))) v4i*)(uint32_t)(uintptr_t)l,
      0, 0);
#else
  *(v8bf*)l = *(const v8bf*)g;
#endif
}

// Wait until at most <n> of this wave's async copies remain in flight.
// Async loads complete in order, so waiting to 4 releases the older tile
// group while the 4 just-issued prefetch copies keep streaming.
__device__ __forceinline__ void wait_async_grp()
{
#if ASYNC_TILE
#  if defined(__has_builtin) && __has_builtin(__builtin_amdgcn_s_wait_asynccnt)
  __builtin_amdgcn_s_wait_asynccnt(4);
#  else
  asm volatile("s_wait_asynccnt 4" ::: "memory");
#  endif
#endif
}
__device__ __forceinline__ void wait_async_all()
{
#if ASYNC_TILE
#  if defined(__has_builtin) && __has_builtin(__builtin_amdgcn_s_wait_asynccnt)
  __builtin_amdgcn_s_wait_asynccnt(0);
#  else
  asm volatile("s_wait_asynccnt 0" ::: "memory");
#  endif
#endif
}

// ---------------------------------------------------------------------------
// f32 -> bf16 conversion (with optional zero tail padding for OOB-safe tiles)
// ---------------------------------------------------------------------------
__global__ __launch_bounds__(256)
void cvt_f32_bf16(const float* __restrict__ in, __bf16* __restrict__ out, int n, int ntot)
{
  const int i = blockIdx.x * 256 + threadIdx.x;
  if (i < ntot) out[i] = (i < n) ? (__bf16)in[i] : (__bf16)0.0f;
}

// ---------------------------------------------------------------------------
// Generic NT GEMM, bf16 WMMA, f32 accumulate, double-buffered async tiles.
//   C[M x N] = A[M x K (lda)] * W[N x K]^T      (A, W already bf16)
// Block: 128 threads (4 waves), tile 64x64, K-step 32. LDS rows padded to
// 48 bf16 (96 B) so every 8-element lane segment is 16-byte aligned.
// epi: 0 = plain, 1 = softplus(c + bias[col]). Cbf: optional bf16 copy out.
// ---------------------------------------------------------------------------
#define BM 64
#define BN 64
#define BK 32
#define LDK 48

__global__ __launch_bounds__(128)
void wmma_gemm_nt(const __bf16* __restrict__ A, const __bf16* __restrict__ W,
                  float* __restrict__ C, __bf16* __restrict__ Cbf,
                  const float* __restrict__ bias,
                  int N, int K, int lda, int ldc, int epi)
{
  __shared__ __bf16 sA[2][BM][LDK];
  __shared__ __bf16 sW[2][BN][LDK];

  const int tid   = threadIdx.x;
  const int lane  = tid & 31;
  const int wave  = tid >> 5;
  const int m0    = blockIdx.y * BM;
  const int n0    = blockIdx.x * BN;
  const int nlane = lane & 15;
  const int khalf = lane >> 4;
  const int am    = (wave << 4) + nlane;  // A-frag row within tile
  const int kg    = khalf * 8;            // A-frag K sub-offset
  const int kh    = khalf * 16;           // unused; B-frag also uses kg

  // Each thread copies 2 x 16B chunks per tile (64 rows x 32 bf16 = 256 chunks).
  const int id0 = tid,        row0 = id0 >> 2, c0 = (id0 & 3) << 3;
  const int id1 = 128 + tid,  row1 = id1 >> 2, c1 = (id1 & 3) << 3;

  auto issue_tiles = [&](int k0, int buf) {
    copy16(A + (size_t)(m0 + row0) * lda + k0 + c0, &sA[buf][row0][c0]);
    copy16(W + (size_t)(n0 + row0) * K   + k0 + c0, &sW[buf][row0][c0]);
    copy16(A + (size_t)(m0 + row1) * lda + k0 + c1, &sA[buf][row1][c1]);
    copy16(W + (size_t)(n0 + row1) * K   + k0 + c1, &sW[buf][row1][c1]);
  };

  v8f acc[4] = {};

  issue_tiles(0, 0);
  int cur = 0;
  for (int k0 = 0; k0 < K; k0 += BK) {
    const bool more = (k0 + BK) < K;
    if (more) issue_tiles(k0 + BK, cur ^ 1);    // prefetch next tile group
    if (more) wait_async_grp();                 // older 4 copies have landed
    else      wait_async_all();
    __syncthreads();

    // A fragment (16x32): row = lane%16; lanes 0-15: K[0..7]+[16..23],
    // lanes 16-31: K[8..15]+[24..31]  -> two ds_load_b128.
    const v8bf a0 = *(const v8bf*)&sA[cur][am][kg];
    const v8bf a1 = *(const v8bf*)&sA[cur][am][16 + kg];
    const v16bf afr = __builtin_shufflevector(a0, a1,
        0, 1, 2, 3, 4, 5, 6, 7, 8, 9, 10, 11, 12, 13, 14, 15);

    #pragma unroll
    for (int t = 0; t < 4; t++) {
      // B fragment (32x16): col = lane%16; lanes 0-15: K[0..7]+[16..23], lanes 16-31: K[8..15]+[24..31].
      const int bn = (t << 4) + nlane;
      const v8bf b0 = *(const v8bf*)&sW[cur][bn][kg];
      const v8bf b1 = *(const v8bf*)&sW[cur][bn][16 + kg];
      const v16bf bfr = __builtin_shufflevector(b0, b1,
          0, 1, 2, 3, 4, 5, 6, 7, 8, 9, 10, 11, 12, 13, 14, 15);
      acc[t] = __builtin_amdgcn_wmma_f32_16x16x32_bf16(
          false, afr, false, bfr, (short)0, acc[t], false, false);
    }
    __syncthreads();   // protect buf 'cur' before it is refilled next+1 step
    cur ^= 1;
  }

  // C/D layout: VGPR r -> M=r (lanes 0-15) / r+8 (lanes 16-31), N=lane%16.
  const int mh = khalf << 3;
  #pragma unroll
  for (int t = 0; t < 4; t++) {
    const int col = n0 + (t << 4) + nlane;
    if (col >= N) continue;
    const float bv = (epi == 1) ? bias[col] : 0.0f;
    #pragma unroll
    for (int r = 0; r < 8; r++) {
      const int row = m0 + (wave << 4) + mh + r;
      float v = acc[t][r];
      if (epi == 1) v = softplusf_(v + bv);
      C[(size_t)row * ldc + col] = v;
      if (Cbf) Cbf[(size_t)row * ldc + col] = (__bf16)v;
    }
  }
}

// ---------------------------------------------------------------------------
// Depthwise causal conv (K=4) + SiLU. Emits f32 (for scan) and bf16 (for GEMM).
// ---------------------------------------------------------------------------
__global__ __launch_bounds__(256)
void conv_silu_kernel(const float* __restrict__ xz, const float* __restrict__ conv_w,
                      float* __restrict__ xc, __bf16* __restrict__ xc_bf)
{
  const int idx = blockIdx.x * 256 + threadIdx.x;     // over B*L*ED
  const int e   = idx & (ED_ - 1);
  const int bl  = idx >> 11;                          // b*L + l
  const int l   = bl & (L_ - 1);
  const float* w = conv_w + e * 4;
  float s = 0.0f;
  #pragma unroll
  for (int k = 0; k < 4; k++) {
    const int ls = l - 3 + k;
    if (ls >= 0) s += xz[(size_t)(bl - 3 + k) * (2 * ED_) + e] * w[k];
  }
  const float v = s * sigmoidf_(s);
  xc[(size_t)bl * ED_ + e]    = v;
  xc_bf[(size_t)bl * ED_ + e] = (__bf16)v;
}

// ---------------------------------------------------------------------------
// Chunked selective scan. Composite chunk decay = exp(A[n] * sum(delta)).
// ---------------------------------------------------------------------------
__global__ __launch_bounds__(256)
void scan_pass1(const float* __restrict__ xc, const float* __restrict__ dBC,
                const float* __restrict__ delta, const float* __restrict__ A_log,
                float* __restrict__ Sout, float* __restrict__ sumd)
{
  const int t  = blockIdx.x;           // B * NC * (ED/256) = 128 blocks
  const int eb = t & 7;
  const int c  = (t >> 3) & (NC_ - 1);
  const int b  = t >> 6;
  const int e  = (eb << 8) + threadIdx.x;

  float Av[NS_];
  #pragma unroll
  for (int n = 0; n < NS_; n++) Av[n] = -__expf(A_log[e * NS_ + n]);
  float h[NS_];
  #pragma unroll
  for (int n = 0; n < NS_; n++) h[n] = 0.0f;

  float sd = 0.0f;
  size_t row = (size_t)b * L_ + (size_t)c * LC_;
  for (int l = 0; l < LC_; l++, row++) {
    const float dlt = delta[row * ED_ + e];
    const float xv  = xc[row * ED_ + e];
    const float* bc = dBC + row * FF_;
    sd += dlt;
    #pragma unroll
    for (int n = 0; n < NS_; n++) {
      const float dA = __expf(dlt * Av[n]);
      h[n] = h[n] * dA + dlt * bc[RR_ + n] * xv;
    }
  }
  const size_t o = (size_t)(b * NC_ + c) * ED_ + e;
  sumd[o] = sd;
  #pragma unroll
  for (int n = 0; n < NS_; n++) Sout[o * NS_ + n] = h[n];
}

__global__ __launch_bounds__(256)
void scan_pass2(const float* __restrict__ Sin, const float* __restrict__ sumd,
                const float* __restrict__ A_log, float* __restrict__ hIn)
{
  const int b = blockIdx.x >> 3;                       // B * (ED/256) = 16 blocks
  const int e = ((blockIdx.x & 7) << 8) + threadIdx.x;

  float Av[NS_];
  #pragma unroll
  for (int n = 0; n < NS_; n++) Av[n] = -__expf(A_log[e * NS_ + n]);
  float h[NS_];
  #pragma unroll
  for (int n = 0; n < NS_; n++) h[n] = 0.0f;

  for (int c = 0; c < NC_; c++) {
    const size_t o = (size_t)(b * NC_ + c) * ED_ + e;
    #pragma unroll
    for (int n = 0; n < NS_; n++) hIn[o * NS_ + n] = h[n];
    const float sd = sumd[o];
    #pragma unroll
    for (int n = 0; n < NS_; n++) {
      const float P = __expf(sd * Av[n]);
      h[n] = h[n] * P + Sin[o * NS_ + n];
    }
  }
}

__global__ __launch_bounds__(256)
void scan_pass3(const float* __restrict__ xz, const float* __restrict__ xc,
                const float* __restrict__ dBC, const float* __restrict__ delta,
                const float* __restrict__ A_log, const float* __restrict__ Dp,
                const float* __restrict__ hIn, __bf16* __restrict__ pre_bf)
{
  const int t  = blockIdx.x;           // B * NC * (ED/256) = 128 blocks
  const int eb = t & 7;
  const int c  = (t >> 3) & (NC_ - 1);
  const int b  = t >> 6;
  const int e  = (eb << 8) + threadIdx.x;

  float Av[NS_];
  #pragma unroll
  for (int n = 0; n < NS_; n++) Av[n] = -__expf(A_log[e * NS_ + n]);
  const float Dv = Dp[e];

  const size_t o = (size_t)(b * NC_ + c) * ED_ + e;
  float h[NS_];
  #pragma unroll
  for (int n = 0; n < NS_; n++) h[n] = hIn[o * NS_ + n];

  size_t row = (size_t)b * L_ + (size_t)c * LC_;
  for (int l = 0; l < LC_; l++, row++) {
    const float dlt = delta[row * ED_ + e];
    const float xv  = xc[row * ED_ + e];
    const float zv  = xz[row * (2 * ED_) + ED_ + e];
    const float* bc = dBC + row * FF_;
    float acc = 0.0f;
    #pragma unroll
    for (int n = 0; n < NS_; n++) {
      const float dA = __expf(dlt * Av[n]);
      h[n] = h[n] * dA + dlt * bc[RR_ + n] * xv;
      acc += h[n] * bc[RR_ + NS_ + n];
    }
    const float y = acc + Dv * xv;
    pre_bf[row * ED_ + e] = (__bf16)(y * (zv * sigmoidf_(zv)));
  }
}

// ---------------------------------------------------------------------------
// Host-side orchestration (graph-capture safe: only kernel launches on stream).
// ---------------------------------------------------------------------------
extern "C" void kernel_launch(void* const* d_in, const int* in_sizes, int n_in,
                              void* d_out, int out_size, void* d_ws, size_t ws_size,
                              hipStream_t stream)
{
  (void)in_sizes; (void)n_in; (void)out_size; (void)ws_size;

  const float* x         = (const float*)d_in[0];   // (B, L, DM)
  const float* in_proj_w = (const float*)d_in[1];   // (2*ED, DM)
  const float* conv_w    = (const float*)d_in[2];   // (ED, 1, 4)
  const float* x_proj_w  = (const float*)d_in[3];   // (96, ED)
  const float* dt_proj_w = (const float*)d_in[4];   // (ED, R)
  const float* dt_proj_b = (const float*)d_in[5];   // (ED,)
  const float* A_log     = (const float*)d_in[6];   // (ED, N)
  const float* Dp        = (const float*)d_in[7];   // (ED,)
  const float* out_proj_w= (const float*)d_in[8];   // (DM, ED)
  float*       out       = (float*)d_out;           // (B, L, DM)

  const int M = B_ * L_;                            // 2048 rows everywhere

  // ---- workspace layout: f32 region, then bf16 region (all 16B aligned) ----
  float* wsf   = (float*)d_ws;
  float* xz    = wsf;                                  // M x 4096
  float* xc    = xz    + (size_t)M * (2 * ED_);        // M x 2048
  float* dBC   = xc    + (size_t)M * ED_;              // M x 96
  float* delta = dBC   + (size_t)M * FF_;              // M x 2048
  float* Schk  = delta + (size_t)M * ED_;              // B*NC x ED x 16
  float* hIn   = Schk  + (size_t)B_ * NC_ * ED_ * NS_; // B*NC x ED x 16
  float* sumd  = hIn   + (size_t)B_ * NC_ * ED_ * NS_; // B*NC x ED
  __bf16* wsb   = (__bf16*)(sumd + (size_t)B_ * NC_ * ED_);
  __bf16* x_bf  = wsb;                                 // M x 1024
  __bf16* ipw_b = x_bf  + (size_t)M * DM_;             // 4096 x 1024
  __bf16* xpw_b = ipw_b + (size_t)(2 * ED_) * DM_;     // 128 x 2048 (zero-padded)
  __bf16* dtw_b = xpw_b + (size_t)FP_ * ED_;           // 2048 x 64
  __bf16* opw_b = dtw_b + (size_t)ED_ * RR_;           // 1024 x 2048
  __bf16* xc_bf = opw_b + (size_t)DM_ * ED_;           // M x 2048
  __bf16* dBC_b = xc_bf + (size_t)M * ED_;             // M x 96
  __bf16* pre_b = dBC_b + (size_t)M * FF_;             // M x 2048

  const dim3 gblk(128);
  #define CVT(src, dst, n, ntot) \
    cvt_f32_bf16<<<((ntot) + 255) / 256, 256, 0, stream>>>((src), (dst), (n), (ntot))

  // 0) one-time-per-call precision demotion of GEMM operands
  CVT(x,          x_bf,  M * DM_,          M * DM_);
  CVT(in_proj_w,  ipw_b, (2 * ED_) * DM_,  (2 * ED_) * DM_);
  CVT(x_proj_w,   xpw_b, FF_ * ED_,        FP_ * ED_);      // pad rows 96..127 = 0
  CVT(dt_proj_w,  dtw_b, ED_ * RR_,        ED_ * RR_);
  CVT(out_proj_w, opw_b, DM_ * ED_,        DM_ * ED_);

  // 1) xz = x @ in_proj_w^T : M=2048, N=4096, K=1024
  wmma_gemm_nt<<<dim3((2 * ED_) / BN, M / BM), gblk, 0, stream>>>(
      x_bf, ipw_b, xz, nullptr, nullptr, 2 * ED_, DM_, DM_, 2 * ED_, 0);

  // 2) xc = silu(causal_conv(xz[:, :ED]))
  conv_silu_kernel<<<(B_ * L_ * ED_) / 256, 256, 0, stream>>>(xz, conv_w, xc, xc_bf);

  // 3) dBC = xc @ x_proj_w^T : N=96 (OOB-safe via padded weights), K=2048
  wmma_gemm_nt<<<dim3((FF_ + BN - 1) / BN, M / BM), gblk, 0, stream>>>(
      xc_bf, xpw_b, dBC, dBC_b, nullptr, FF_, ED_, ED_, FF_, 0);

  // 4) delta = softplus(dBC[:, :R] @ dt_proj_w^T + b) : N=2048, K=64, lda=96
  wmma_gemm_nt<<<dim3(ED_ / BN, M / BM), gblk, 0, stream>>>(
      dBC_b, dtw_b, delta, nullptr, dt_proj_b, ED_, RR_, FF_, ED_, 1);

  // 5) chunked selective scan + D skip + silu(z) gating -> pre (bf16)
  scan_pass1<<<B_ * NC_ * (ED_ / 256), 256, 0, stream>>>(xc, dBC, delta, A_log, Schk, sumd);
  scan_pass2<<<B_ * (ED_ / 256), 256, 0, stream>>>(Schk, sumd, A_log, hIn);
  scan_pass3<<<B_ * NC_ * (ED_ / 256), 256, 0, stream>>>(xz, xc, dBC, delta, A_log, Dp, hIn, pre_b);

  // 6) out = pre @ out_proj_w^T : N=1024, K=2048
  wmma_gemm_nt<<<dim3(DM_ / BN, M / BM), gblk, 0, stream>>>(
      pre_b, opw_b, out, nullptr, nullptr, DM_, ED_, ED_, DM_, 0);
  #undef CVT
}
